// SeparableConvolution_50852412785139
// MI455X (gfx1250) — matrix operand order repaired and checked by reading, not compiled, not run on hardware
//
#include <hip/hip_runtime.h>
#include <hip/hip_bf16.h>

#define NN 100000
#define NE 3200000
#define MT 2                       // node-tiles (of 16) per wave

typedef __attribute__((ext_vector_type(16))) __bf16 v16bf;
typedef __attribute__((ext_vector_type(8)))  __bf16 v8bf;
typedef __attribute__((ext_vector_type(8)))  float  v8f;

__device__ inline __bf16 f2bf(float f) {
  unsigned u = __builtin_bit_cast(unsigned, f);
  u += 0x7FFFu + ((u >> 16) & 1u);          // round-to-nearest-even
  unsigned short h = (unsigned short)(u >> 16);
  return __builtin_bit_cast(__bf16, h);
}

// ---------------------------------------------------------------------------
// Pre-swizzle a weight matrix into WMMA B-fragment order (bf16).
// Flat bf16 index: ((t*2+s)*32 + lane)*16 + e
// B element (K = s*32 + (e<8?0:16) + (lane<16?0:8) + (e&7), col = t*16 + (lane&15))
// mode 0: W is (64 x ntiles*16) row-major   (kernel-MLP W2)
// mode 1: W is (out x in) row-major, use W^T (mix_W)
// ---------------------------------------------------------------------------
__global__ void preswizzle_kernel(const float* __restrict__ W, __bf16* __restrict__ out,
                                  int ntiles, int mode) {
  int idx = blockIdx.x * blockDim.x + threadIdx.x;
  int total = ntiles * 1024;
  if (idx >= total) return;
  int e = idx & 15;
  int l = (idx >> 4) & 31;
  int s = (idx >> 9) & 1;
  int t = idx >> 10;
  int K = s * 32 + ((e >> 3) << 4) + ((l >> 4) << 3) + (e & 7);
  int col = t * 16 + (l & 15);
  float v = (mode == 0) ? W[K * (ntiles * 16) + col] : W[col * 64 + K];
  out[idx] = f2bf(v);
}

__global__ void zero_kernel(float* __restrict__ p, int n) {
  int i = blockIdx.x * blockDim.x + threadIdx.x;
  if (i < n) p[i] = 0.0f;
}

// --------------------- CSR inversion (replaces 205M f32 atomics) -----------
__global__ void count_kernel(const int* __restrict__ ei, int* __restrict__ deg) {
  int e = blockIdx.x * blockDim.x + threadIdx.x;
  if (e < NE) atomicAdd(&deg[ei[e]], 1);          // ei[0..E) = dst
}

__global__ void reserve_kernel(const int* __restrict__ deg, int* __restrict__ start,
                               int* __restrict__ cursor, int* __restrict__ gcur) {
  int n = blockIdx.x * blockDim.x + threadIdx.x;
  if (n < NN) {
    int s = atomicAdd(gcur, deg[n]);   // contiguous region per node; order irrelevant
    start[n] = s;
    cursor[n] = s;
  }
}

__global__ void fill_kernel(const int* __restrict__ ei, int* __restrict__ cursor,
                            int* __restrict__ csr) {
  int e = blockIdx.x * blockDim.x + threadIdx.x;
  if (e < NE) {
    int dst = ei[e];
    int src = ei[NE + e];
    int p = atomicAdd(&cursor[dst], 1);
    csr[p] = src;
  }
}

// 64 threads per node: each thread owns one column, streams neighbor rows.
__global__ __launch_bounds__(256)
void gather_kernel(const float* __restrict__ kyv, const int* __restrict__ csr,
                   const int* __restrict__ start, const int* __restrict__ deg,
                   float* __restrict__ mean) {
  int node = blockIdx.x * 4 + (threadIdx.x >> 6);
  int col = threadIdx.x & 63;
  if (node >= NN) return;
  int s = start[node];
  int d = deg[node];
  float acc = 0.0f;
#pragma unroll 4
  for (int i = 0; i < d; ++i)
    acc += kyv[(long)csr[s + i] * 64 + col];
  mean[(long)node * 64 + col] = acc / fmaxf((float)d, 1.0f);   // scatter_mean
}

// ---------------------------------------------------------------------------
// Fused per-node pipeline. One wave (32 lanes) per MT*16 nodes:
//   h   = relu(ea @ W1 + b1)                 (VALU, K=3)
//   w   = h @ W2 + b2 tile-by-tile           (v_wmma_f32_16x16x32_bf16)
//   y   = block_matmul(x, w)                 (8 VALU FMAs per tile per m)
//   [FUSE_MIX] out = y @ mix_W^T + mix_b     (2 WMMAs/tile, bias in C)
// Each B fragment is reused MT times.
// ---------------------------------------------------------------------------
template <bool FUSE_MIX>
__global__ __launch_bounds__(32)
void node_gemm_kernel(const float* __restrict__ xin,
                      const float* __restrict__ ea,
                      const float* __restrict__ W1,
                      const float* __restrict__ b1,
                      const v16bf* __restrict__ Bfrag,
                      const float* __restrict__ b2,
                      const v16bf* __restrict__ Mfrag,
                      const float* __restrict__ mix_b,
                      float* __restrict__ outp) {
  __shared__ __align__(16) float  xsh[MT * 1024];
  __shared__ __align__(16) __bf16 hsh[MT * 1024];
  __shared__ __align__(16) float  b2sh[1024];
  __shared__ __align__(16) __bf16 vsh[MT * 1024];

  const int lane = threadIdx.x;
  const int hi = lane >> 4;
  const int lj = lane & 15;
  const int node0 = blockIdx.x * (16 * MT);   // N divisible by 32

  // Stage x and h (bf16) into LDS: each lane covers MT*32 of MT*1024 values.
#pragma unroll
  for (int i = 0; i < MT * 32; ++i) {
    int v = lane + 32 * i;
    int nd = v >> 6;
    int k = v & 63;
    xsh[v] = xin[(node0 + nd) * 64 + k];
    float e0 = ea[(node0 + nd) * 3 + 0];
    float e1 = ea[(node0 + nd) * 3 + 1];
    float e2 = ea[(node0 + nd) * 3 + 2];
    float h = fmaf(e0, W1[k], fmaf(e1, W1[64 + k], fmaf(e2, W1[128 + k], b1[k])));
    hsh[v] = f2bf(fmaxf(h, 0.0f));
  }
#pragma unroll
  for (int i = 0; i < 32; ++i) b2sh[lane + 32 * i] = b2[lane + 32 * i];
  __syncthreads();

  // A fragments (16x32 bf16): lane holds row M=lj; K = (e<8?0:16)+hi*8+(e&7)
  v16bf a0[MT], a1[MT];
#pragma unroll
  for (int m = 0; m < MT; ++m) {
    const v8bf* hrow = (const v8bf*)(hsh + m * 1024 + lj * 64);
    v8bf g0 = hrow[hi];
    v8bf g1 = hrow[2 + hi];
    v8bf g2 = hrow[4 + hi];
    v8bf g3 = hrow[6 + hi];
    a0[m] = __builtin_shufflevector(g0, g1, 0,1,2,3,4,5,6,7,8,9,10,11,12,13,14,15);
    a1[m] = __builtin_shufflevector(g2, g3, 0,1,2,3,4,5,6,7,8,9,10,11,12,13,14,15);
  }

  v8f yacc[MT][4] = {};
#pragma unroll
  for (int c = 0; c < 4; ++c) {
#pragma unroll
    for (int k = 0; k < 16; ++k) {
      const int t = c * 16 + k;            // N-tile of (16 x 64) @ (64 x 1024)
      v16bf bf0 = Bfrag[(t * 2 + 0) * 32 + lane];
      v16bf bf1 = Bfrag[(t * 2 + 1) * 32 + lane];
      const float bias = b2sh[t * 16 + lj];
#pragma unroll
      for (int m = 0; m < MT; ++m) {
        v8f acc = {};
        acc = __builtin_amdgcn_wmma_f32_16x16x32_bf16(false, a0[m], false, bf0,
                                                      (short)0, acc, false, false);
        acc = __builtin_amdgcn_wmma_f32_16x16x32_bf16(false, a1[m], false, bf1,
                                                      (short)0, acc, false, false);
        // block-diagonal matmul: y[n, c*16+j] += x[n, c*16+k] * (w + b2)
#pragma unroll
        for (int r = 0; r < 8; ++r) {
          float xv = xsh[m * 1024 + (r + 8 * hi) * 64 + c * 16 + k];
          yacc[m][c][r] = fmaf(acc[r] + bias, xv, yacc[m][c][r]);
        }
      }
    }
  }

  if constexpr (!FUSE_MIX) {
#pragma unroll
    for (int m = 0; m < MT; ++m)
#pragma unroll
      for (int c = 0; c < 4; ++c)
#pragma unroll
        for (int r = 0; r < 8; ++r)
          outp[(node0 + m * 16 + r + 8 * hi) * 64 + c * 16 + lj] = yacc[m][c][r];
  } else {
    // round-trip y through LDS as bf16 to rebuild A fragments for the mix GEMM
#pragma unroll
    for (int m = 0; m < MT; ++m)
#pragma unroll
      for (int c = 0; c < 4; ++c)
#pragma unroll
        for (int r = 0; r < 8; ++r)
          vsh[m * 1024 + (r + 8 * hi) * 64 + c * 16 + lj] = f2bf(yacc[m][c][r]);
    __syncthreads();
    v16bf va0[MT], va1[MT];
#pragma unroll
    for (int m = 0; m < MT; ++m) {
      const v8bf* vrow = (const v8bf*)(vsh + m * 1024 + lj * 64);
      v8bf q0 = vrow[hi];
      v8bf q1 = vrow[2 + hi];
      v8bf q2 = vrow[4 + hi];
      v8bf q3 = vrow[6 + hi];
      va0[m] = __builtin_shufflevector(q0, q1, 0,1,2,3,4,5,6,7,8,9,10,11,12,13,14,15);
      va1[m] = __builtin_shufflevector(q2, q3, 0,1,2,3,4,5,6,7,8,9,10,11,12,13,14,15);
    }
#pragma unroll
    for (int t2 = 0; t2 < 4; ++t2) {
      float mb = mix_b[t2 * 16 + lj];
      v16bf m0 = Mfrag[(t2 * 2 + 0) * 32 + lane];
      v16bf m1 = Mfrag[(t2 * 2 + 1) * 32 + lane];
#pragma unroll
      for (int m = 0; m < MT; ++m) {
        v8f macc;
#pragma unroll
        for (int r = 0; r < 8; ++r) macc[r] = mb;   // bias pre-loaded into C
        macc = __builtin_amdgcn_wmma_f32_16x16x32_bf16(false, va0[m], false, m0,
                                                       (short)0, macc, false, false);
        macc = __builtin_amdgcn_wmma_f32_16x16x32_bf16(false, va1[m], false, m1,
                                                       (short)0, macc, false, false);
#pragma unroll
        for (int r = 0; r < 8; ++r)
          outp[(node0 + m * 16 + r + 8 * hi) * 64 + t2 * 16 + lj] = macc[r];
      }
    }
  }
}

extern "C" void kernel_launch(void* const* d_in, const int* in_sizes, int n_in,
                              void* d_out, int out_size, void* d_ws, size_t ws_size,
                              hipStream_t stream) {
  const float* x     = (const float*)d_in[0];
  const float* ea    = (const float*)d_in[1];
  const int*   eidx  = (const int*)d_in[2];
  const float* k1_W1 = (const float*)d_in[3];
  const float* k1_b1 = (const float*)d_in[4];
  const float* k1_W2 = (const float*)d_in[5];
  const float* k1_b2 = (const float*)d_in[6];
  const float* k2_W1 = (const float*)d_in[7];
  const float* k2_b1 = (const float*)d_in[8];
  const float* k2_W2 = (const float*)d_in[9];
  const float* k2_b2 = (const float*)d_in[10];
  const float* mix_W = (const float*)d_in[11];
  const float* mix_b = (const float*)d_in[12];
  float* out = (float*)d_out;

  // workspace layout (fragment regions 32B-aligned)
  char* ws = (char*)d_ws;
  __bf16* B1   = (__bf16*)(ws + 0);                   // 128 KB
  __bf16* B2   = (__bf16*)(ws + 131072);              // 128 KB
  __bf16* B3   = (__bf16*)(ws + 262144);              // 8 KB
  float* kyv   = (float*)(ws + 270336);               // N*64*4 = 25.6 MB
  float* mean  = (float*)(ws + 25870336);             // N*64*4 = 25.6 MB
  int*   deg   = (int*)  (ws + 51470336);             // N*4
  int*   gcur  = (int*)  (ws + 51870336);             // 64 B (contiguous after deg)
  int*   start = (int*)  (ws + 51870400);             // N*4
  int*   curs  = (int*)  (ws + 52270400);             // N*4
  int*   csr   = (int*)  (ws + 52670400);             // E*4 = 12.8 MB

  // 1) zero degree counts + global cursor (contiguous region, bits(0.0f)==0)
  zero_kernel<<<(NN + 16 + 255) / 256, 256, 0, stream>>>((float*)deg, NN + 16);

  // 2) pre-swizzle weights into WMMA B-fragment layout
  preswizzle_kernel<<<(65536 + 255) / 256, 256, 0, stream>>>(k1_W2, B1, 64, 0);
  preswizzle_kernel<<<(65536 + 255) / 256, 256, 0, stream>>>(k2_W2, B2, 64, 0);
  preswizzle_kernel<<<(4096 + 255) / 256, 256, 0, stream>>>(mix_W, B3, 4, 1);

  // 3) ky_v = block_matmul(x, kernel1(edge_attr))
  node_gemm_kernel<false><<<NN / (16 * MT), 32, 0, stream>>>(
      x, ea, k1_W1, k1_b1, (const v16bf*)B1, k1_b2, nullptr, nullptr, kyv);

  // 4) CSR inversion + mean gather (atomics: 6.5M int vs 205M f32)
  count_kernel<<<(NE + 255) / 256, 256, 0, stream>>>(eidx, deg);
  reserve_kernel<<<(NN + 255) / 256, 256, 0, stream>>>(deg, start, curs, gcur);
  fill_kernel<<<(NE + 255) / 256, 256, 0, stream>>>(eidx, curs, csr);
  gather_kernel<<<(NN + 3) / 4, 256, 0, stream>>>(kyv, csr, start, deg, mean);

  // 5) v = block_matmul(mean, kernel2(edge_attr)); out = v @ mix_W^T + mix_b
  node_gemm_kernel<true><<<NN / (16 * MT), 32, 0, stream>>>(
      mean, ea, k2_W1, k2_b1, (const v16bf*)B2, k2_b2,
      (const v16bf*)B3, mix_b, out);
}
